// SAGEBlock_42348377538964
// MI455X (gfx1250) — matrix operand order repaired and checked by reading, not compiled, not run on hardware
//
#include <hip/hip_runtime.h>
#include <hip/hip_bf16.h>
#include <math.h>

#define FD 256  // feature dim

typedef __attribute__((ext_vector_type(16))) __bf16 v16bf;
typedef __attribute__((ext_vector_type(8)))  __bf16 v8bf;
typedef __attribute__((ext_vector_type(8)))  float  v8f;

// ---------- native bf16 split: hi = RNE(a), lo = RNE(a - hi) ----------
__device__ __forceinline__ void split1(float a, __bf16& h, __bf16& l) {
  __bf16 hh = (__bf16)a;          // v_cvt_pk_bf16_f32 class (RNE is HW default)
  h = hh;
  l = (__bf16)(a - (float)hh);    // residual
}

__device__ __forceinline__ v16bf cat8(v8bf a, v8bf b) {
  return __builtin_shufflevector(a, b, 0,1,2,3,4,5,6,7,8,9,10,11,12,13,14,15);
}

// A fragment from pre-split bf16 plane: two 8-elem (16B, b128-aligned) runs.
__device__ __forceinline__ v16bf load_a_pre(const __bf16* p) {
  return cat8(*(const v8bf*)p, *(const v8bf*)(p + 8));
}

// Fallback: load two contiguous 8-float runs, scale, split into bf16 hi/lo.
__device__ __forceinline__ void load_split_a(const float* p0, const float* p1,
                                             float scale, v16bf& hi, v16bf& lo) {
  float v[16];
  float4 t;
  t = *(const float4*)(p0 + 0); v[0]=t.x;  v[1]=t.y;  v[2]=t.z;  v[3]=t.w;
  t = *(const float4*)(p0 + 4); v[4]=t.x;  v[5]=t.y;  v[6]=t.z;  v[7]=t.w;
  t = *(const float4*)(p1 + 0); v[8]=t.x;  v[9]=t.y;  v[10]=t.z; v[11]=t.w;
  t = *(const float4*)(p1 + 4); v[12]=t.x; v[13]=t.y; v[14]=t.z; v[15]=t.w;
#pragma unroll
  for (int i = 0; i < 16; ++i) {
    __bf16 h, l;
    split1(v[i] * scale, h, l);
    hi[i] = h;
    lo[i] = l;
  }
}

// ---------- kernel 0: zero msg accumulator + degree ----------
__global__ void sage_zero_kernel(float* __restrict__ p, long long count) {
  long long i = (long long)blockIdx.x * blockDim.x + threadIdx.x;
  long long stride = (long long)gridDim.x * blockDim.x;
  for (; i < count; i += stride) p[i] = 0.0f;
}

// ---------- kernel 1: split both weight matrices into bf16 hi/lo planes ----------
__global__ void __launch_bounds__(256)
sage_wsplit_kernel(const float* __restrict__ Wl, const float* __restrict__ Wr,
                   __bf16* __restrict__ wlhi, __bf16* __restrict__ wllo,
                   __bf16* __restrict__ wrhi, __bf16* __restrict__ wrlo) {
  const int i = blockIdx.x * blockDim.x + threadIdx.x;
  if (i < FD * FD) {
    __bf16 h, l;
    split1(Wl[i], h, l); wlhi[i] = h; wllo[i] = l;
    split1(Wr[i], h, l); wrhi[i] = h; wrlo[i] = l;
  }
}

// ---------- kernel 2: edge-parallel scatter (mean numerator + degree) ----------
// One wave32 per edge; two b128 gathers + 8 contiguous f32 atomics per lane.
// x and msg are both L2-resident (102.4 MB each vs 192 MB L2).
__global__ void __launch_bounds__(256)
sage_scatter_kernel(const float* __restrict__ x, const int* __restrict__ src,
                    const int* __restrict__ dst, float* __restrict__ msg,
                    float* __restrict__ deg, int e)
{
  const int gid  = blockIdx.x * blockDim.x + threadIdx.x;
  const int edge = gid >> 5;
  const int lane = gid & 31;
  if (edge >= e) return;
  const int s = src[edge];
  const int d = dst[edge];
  const float* xs = x   + (size_t)s * FD;
  float*       md = msg + (size_t)d * FD;
  if (lane == 0) atomicAdd(deg + d, 1.0f);
  const float4 a0 = *(const float4*)(xs + 4 * lane);
  const float4 a1 = *(const float4*)(xs + 128 + 4 * lane);
  float* p0 = md + 4 * lane;
  float* p1 = md + 128 + 4 * lane;
  atomicAdd(p0 + 0, a0.x); atomicAdd(p0 + 1, a0.y);
  atomicAdd(p0 + 2, a0.z); atomicAdd(p0 + 3, a0.w);
  atomicAdd(p1 + 0, a1.x); atomicAdd(p1 + 1, a1.y);
  atomicAdd(p1 + 2, a1.z); atomicAdd(p1 + 3, a1.w);
}

// ---------- kernel 3: split activations: agg = msg/deg and x into bf16 hi/lo ----------
__global__ void __launch_bounds__(256)
sage_asplit_kernel(const float* __restrict__ msg, const float* __restrict__ deg,
                   const float* __restrict__ x,
                   __bf16* __restrict__ aghi, __bf16* __restrict__ aglo,
                   __bf16* __restrict__ xhi,  __bf16* __restrict__ xlo,
                   long long count)
{
  long long i = (long long)blockIdx.x * blockDim.x + threadIdx.x;
  long long stride = (long long)gridDim.x * blockDim.x;
  for (; i < count; i += stride) {
    const float s = 1.0f / fmaxf(deg[i >> 8], 1.0f);
    __bf16 h, l;
    split1(msg[i] * s, h, l); aghi[i] = h; aglo[i] = l;
    split1(x[i],       h, l); xhi[i]  = h; xlo[i]  = l;
  }
}

// ---------- kernel 4: fused split-bf16 WMMA GEMM + bias + GELU + LN + residual ----
// Workgroup = 128 threads = 4 waves; 32 node rows x 256 output cols.
// Wave w owns cols [64w, 64w+64) for 2 row-tiles -> each B fragment feeds 6 WMMAs.
// Effective K = 512 ([agg | x] vs [Wl^T ; Wr^T]); split-bf16 = 3 WMMA per K=32.
template <bool PRE>
__global__ void __launch_bounds__(128)
sage_fused_kernel(const float* __restrict__ msg, const float* __restrict__ deg,
                  const float* __restrict__ x,
                  const __bf16* __restrict__ aghi, const __bf16* __restrict__ aglo,
                  const __bf16* __restrict__ xhi,  const __bf16* __restrict__ xlo,
                  const __bf16* __restrict__ wlhi, const __bf16* __restrict__ wllo,
                  const __bf16* __restrict__ wrhi, const __bf16* __restrict__ wrlo,
                  const float* __restrict__ bl,
                  const float* __restrict__ gamma, const float* __restrict__ beta,
                  float* __restrict__ out, int n)
{
  __shared__ float sf[32 * FD];   // 32 KB post-GELU staging for LayerNorm
  __shared__ float smu[32];
  __shared__ float srs[32];

  const int lane  = threadIdx.x & 31;
  const int wave  = threadIdx.x >> 5;
  const int rowl  = lane & 15;
  const int kh8   = (lane >> 4) << 3;   // 0 or 8, unused (A also uses kh16)
  const int kh16  = (lane >> 4) << 4;   // B-fragment K sub-offset: 0 or 16
  const int rbase = blockIdx.x * 32;
  int row0 = rbase + rowl;      if (row0 >= n) row0 = n - 1;
  int row1 = rbase + 16 + rowl; if (row1 >= n) row1 = n - 1;
  const int colbase = wave * 64;

  const float s0 = 1.0f / fmaxf(deg[row0], 1.0f);
  const float s1 = 1.0f / fmaxf(deg[row1], 1.0f);

  v8f acc0[4], acc1[4];
  const v8f vzero = {0.f,0.f,0.f,0.f,0.f,0.f,0.f,0.f};
#pragma unroll
  for (int t = 0; t < 4; ++t) { acc0[t] = vzero; acc1[t] = vzero; }

  // phase 0: A = agg (= msg/deg), W = W_l ; phase 1: A = x, W = W_r
  for (int phase = 0; phase < 2; ++phase) {
    const __bf16* whi = phase ? wrhi : wlhi;
    const __bf16* wlo = phase ? wrlo : wllo;
    for (int k0 = 0; k0 < FD; k0 += 32) {
      v16bf ah0, al0, ah1, al1;
      if (PRE) {
        const __bf16* hb = phase ? xhi : aghi;
        const __bf16* lb = phase ? xlo : aglo;
        const size_t o0 = (size_t)row0 * FD + k0 + kh16;
        const size_t o1 = (size_t)row1 * FD + k0 + kh16;
        ah0 = load_a_pre(hb + o0); al0 = load_a_pre(lb + o0);
        ah1 = load_a_pre(hb + o1); al1 = load_a_pre(lb + o1);
      } else {
        const float* A = phase ? x : msg;
        const float sc0 = phase ? 1.0f : s0;
        const float sc1 = phase ? 1.0f : s1;
        const float* ar0 = A + (size_t)row0 * FD + k0 + kh16;
        const float* ar1 = A + (size_t)row1 * FD + k0 + kh16;
        load_split_a(ar0, ar0 + 8, sc0, ah0, al0);
        load_split_a(ar1, ar1 + 8, sc1, ah1, al1);
      }
#pragma unroll
      for (int t = 0; t < 4; ++t) {
        const size_t woff = (size_t)(colbase + t * 16 + rowl) * FD + k0 + kh16;
        const v16bf bh  = *(const v16bf*)(whi + woff);   // 32B packed bf16, aligned
        const v16bf blo = *(const v16bf*)(wlo + woff);
        // split-precision fp32 emulation: hi*hi + hi*lo + lo*hi, fp32 accum
        acc0[t] = __builtin_amdgcn_wmma_f32_16x16x32_bf16(false, ah0, false, bh,
                                                          (short)0, acc0[t], false, false);
        acc0[t] = __builtin_amdgcn_wmma_f32_16x16x32_bf16(false, ah0, false, blo,
                                                          (short)0, acc0[t], false, false);
        acc0[t] = __builtin_amdgcn_wmma_f32_16x16x32_bf16(false, al0, false, bh,
                                                          (short)0, acc0[t], false, false);
        acc1[t] = __builtin_amdgcn_wmma_f32_16x16x32_bf16(false, ah1, false, bh,
                                                          (short)0, acc1[t], false, false);
        acc1[t] = __builtin_amdgcn_wmma_f32_16x16x32_bf16(false, ah1, false, blo,
                                                          (short)0, acc1[t], false, false);
        acc1[t] = __builtin_amdgcn_wmma_f32_16x16x32_bf16(false, al1, false, bh,
                                                          (short)0, acc1[t], false, false);
      }
    }
  }

  // ---- epilogue: bias + exact-erf GELU into LDS (C tile: vgpr i -> row mbase+i, col lane%16)
  const int mbase = (lane >> 4) << 3;
#pragma unroll
  for (int t = 0; t < 4; ++t) {
    const int col = colbase + t * 16 + rowl;
    const float b = bl[col];
#pragma unroll
    for (int i = 0; i < 8; ++i) {
      float f0 = acc0[t][i] + b;
      f0 = 0.5f * f0 * (1.0f + erff(f0 * 0.70710678118654752f));
      sf[(mbase + i) * FD + col] = f0;
      float f1 = acc1[t][i] + b;
      f1 = 0.5f * f1 * (1.0f + erff(f1 * 0.70710678118654752f));
      sf[(16 + mbase + i) * FD + col] = f1;
    }
  }
  __syncthreads();

  // ---- per-row mean / rstd (32 rows, D=256)
  if (threadIdx.x < 32) {
    const float* rp = &sf[threadIdx.x * FD];
    float m = 0.f;
    for (int j = 0; j < FD; ++j) m += rp[j];
    m *= (1.0f / FD);
    float v = 0.f;
    for (int j = 0; j < FD; ++j) { float d = rp[j] - m; v += d * d; }
    v *= (1.0f / FD);
    smu[threadIdx.x] = m;
    srs[threadIdx.x] = rsqrtf(v + 1e-5f);
  }
  __syncthreads();

  // ---- normalize + affine + residual, coalesced store
  for (int idx = threadIdx.x; idx < 32 * FD; idx += 128) {
    const int r = idx >> 8;
    const int c = idx & (FD - 1);
    const int orow = rbase + r;
    if (orow < n) {
      float f = (sf[idx] - smu[r]) * srs[r] * gamma[c] + beta[c];
      out[(size_t)orow * FD + c] = f + x[(size_t)orow * FD + c];
    }
  }
}

extern "C" void kernel_launch(void* const* d_in, const int* in_sizes, int n_in,
                              void* d_out, int out_size, void* d_ws, size_t ws_size,
                              hipStream_t stream) {
  const float* x     = (const float*)d_in[0];
  const int*   ei    = (const int*)d_in[1];   // [2, E] int32 (JAX x64 off)
  const float* Wl    = (const float*)d_in[2];
  const float* bl    = (const float*)d_in[3];
  const float* Wr    = (const float*)d_in[4];
  const float* gamma = (const float*)d_in[5];
  const float* beta  = (const float*)d_in[6];
  float*       out   = (float*)d_out;

  const int n = in_sizes[0] / FD;   // 100000
  const int e = in_sizes[1] / 2;    // 3200000

  // ws layout: [msg: N*256 f32][deg: N f32][4 weight bf16 planes][4 activation bf16 planes]
  float*  msg  = (float*)d_ws;
  float*  deg  = msg + (size_t)n * FD;
  __bf16* wlhi = (__bf16*)(deg + n);              // 16B-aligned offsets throughout
  __bf16* wllo = wlhi + (size_t)FD * FD;
  __bf16* wrhi = wllo + (size_t)FD * FD;
  __bf16* wrlo = wrhi + (size_t)FD * FD;
  __bf16* aghi = wrlo + (size_t)FD * FD;
  __bf16* aglo = aghi + (size_t)n * FD;
  __bf16* xhi  = aglo + (size_t)n * FD;
  __bf16* xlo  = xhi  + (size_t)n * FD;

  const size_t need_full = (size_t)n * FD * 4 + (size_t)n * 4 +
                           4 * (size_t)FD * FD * 2 + 4 * (size_t)n * FD * 2;
  const bool pre = (ws_size >= need_full);

  const long long zcount = (long long)n * (FD + 1);
  sage_zero_kernel<<<2048, 256, 0, stream>>>(msg, zcount);

  sage_wsplit_kernel<<<(FD * FD + 255) / 256, 256, 0, stream>>>(Wl, Wr, wlhi, wllo,
                                                                wrhi, wrlo);

  sage_scatter_kernel<<<(e + 7) / 8, 256, 0, stream>>>(x, ei, ei + e, msg, deg, e);

  const int blocks = (n + 31) / 32;
  if (pre) {
    sage_asplit_kernel<<<8192, 256, 0, stream>>>(msg, deg, x, aghi, aglo, xhi, xlo,
                                                 (long long)n * FD);
    sage_fused_kernel<true><<<blocks, 128, 0, stream>>>(msg, deg, x,
                                                        aghi, aglo, xhi, xlo,
                                                        wlhi, wllo, wrhi, wrlo,
                                                        bl, gamma, beta, out, n);
  } else {
    sage_fused_kernel<false><<<blocks, 128, 0, stream>>>(msg, deg, x,
                                                         aghi, aglo, xhi, xlo,
                                                         wlhi, wllo, wrhi, wrlo,
                                                         bl, gamma, beta, out, n);
  }
}
